// LanguageGatedBundleMatchModule_51934744543994
// MI455X (gfx1250) — matrix operand order repaired and checked by reading, not compiled, not run on hardware
//
#include <hip/hip_runtime.h>
#include <hip/hip_bf16.h>
#include <math.h>
#include <stdint.h>

// ---------------- problem constants (match reference) ----------------
#define Bc   16
#define Nc   2048
#define Kc   16
#define LANGc 256
#define FEATc 128
#define HIDc  128
#define HHc   128
#define ALPHAc 0.5f
#define LN_EPSc 1e-5f
#define PI_Fc 3.14159265358979323846f

typedef __attribute__((ext_vector_type(16))) __bf16 v16bf;
typedef __attribute__((ext_vector_type(8)))  float  v8f;
typedef __attribute__((ext_vector_type(8))) __bf16 v8bf;

// ---------------- WMMA helpers (CDNA5, wave32) ----------------
__device__ __forceinline__ v8f wmma_bf16(v16bf a, v16bf b, v8f c) {
    // (neg_a, A, neg_b, B, c_mod, C, reuse_a, reuse_b)
    return __builtin_amdgcn_wmma_f32_16x16x32_bf16(false, a, false, b, (short)0, c, false, false);
}

// B operand (32x16 K-major tile) from a pre-transposed bf16 weight copy:
// wT[n*Krows + k].  Lane l holds column n=l&15, K rows (l>>4)*8 + [0..7] and 16 + (l>>4)*8 + [0..7].
__device__ __forceinline__ v16bf load_wT_b(const __bf16* wT, int Krows, int s, int ct, int lane) {
    const __bf16* p = wT + (size_t)(ct * 16 + (lane & 15)) * Krows + s * 32 + ((lane >> 4) << 3);
    v8bf lo = *reinterpret_cast<const v8bf*>(p), hi8 = *reinterpret_cast<const v8bf*>(p + 16);
    return __builtin_shufflevector(lo, hi8, 0, 1, 2, 3, 4, 5, 6, 7, 8, 9, 10, 11, 12, 13, 14, 15);  // two 16B runs
}

// A operand channels for lane l, k-step s: runs [c0..c0+7] and [c0+16..c0+23]
__device__ __forceinline__ int a_c0(int s, int lane) { return s * 32 + ((lane >> 4) << 3); }

// ---------------- objectness argmax ----------------
__global__ void k_objn(const float* __restrict__ obj, float* __restrict__ objn) {
    int i = blockIdx.x * 256 + threadIdx.x;
    if (i < Bc * Nc) {
        float a = obj[2 * i], b = obj[2 * i + 1];
        objn[i] = (b > a) ? 1.0f : 0.0f;
    }
}

// ---------------- brute-force KNN (top-17, drop self) ----------------
__global__ void k_knn(const float* __restrict__ xyz, int* __restrict__ knn) {
    __shared__ float sx[256], sy[256], sz[256];
    int b = blockIdx.y;
    int n = blockIdx.x * 256 + threadIdx.x;
    const float* base = xyz + (size_t)b * Nc * 3;
    float qx = base[n * 3 + 0], qy = base[n * 3 + 1], qz = base[n * 3 + 2];
    float bd[Kc + 1];
    int   bi[Kc + 1];
#pragma unroll
    for (int i = 0; i <= Kc; i++) { bd[i] = 3.4e38f; bi[i] = 0; }
    for (int t = 0; t < Nc; t += 256) {
        __syncthreads();
        int j = t + threadIdx.x;
        sx[threadIdx.x] = base[j * 3 + 0];
        sy[threadIdx.x] = base[j * 3 + 1];
        sz[threadIdx.x] = base[j * 3 + 2];
        __syncthreads();
        for (int jj = 0; jj < 256; jj++) {
            float dx = sx[jj] - qx, dy = sy[jj] - qy, dz = sz[jj] - qz;
            float d = dx * dx + dy * dy + dz * dz;
            if (d < bd[Kc]) {
                bd[Kc] = d; bi[Kc] = t + jj;
#pragma unroll
                for (int q = Kc; q > 0; q--) {
                    if (bd[q] < bd[q - 1]) {
                        float td = bd[q]; bd[q] = bd[q - 1]; bd[q - 1] = td;
                        int   ti = bi[q]; bi[q] = bi[q - 1]; bi[q - 1] = ti;
                    }
                }
            }
        }
    }
    int* out = knn + ((size_t)b * Nc + n) * Kc;
#pragma unroll
    for (int i = 0; i < Kc; i++) out[i] = bi[i + 1];  // drop nearest (self)
}

// ---------------- lang_part = lang @ theta_w1[3:] + theta_b1 ----------------
__global__ void k_langpart(const float* __restrict__ lang, const float* __restrict__ w1,
                           const float* __restrict__ b1, float* __restrict__ lp) {
    int b = blockIdx.x, h = threadIdx.x;  // 128 threads
    float acc = b1[h];
    for (int i = 0; i < LANGc; i++) acc = fmaf(lang[b * LANGc + i], w1[(3 + i) * HHc + h], acc);
    lp[b * HHc + h] = acc;
}

// ---------------- weight prep: f32 (rows x cols) -> bf16 transposed ----------------
__global__ void k_prep_wT(const float* __restrict__ w, __bf16* __restrict__ wT, int rows, int cols) {
    int i = blockIdx.x * 256 + threadIdx.x;
    if (i < rows * cols) {
        int r = i / cols, c = i - r * cols;
        wT[(size_t)c * rows + r] = (__bf16)w[i];
    }
}

// ---------------- theta MLP: 128x128 GEMM via WMMA ----------------
// Block = 256 threads (8 waves), 2 points per wave -> 16 points per block.
// w2T (32KB bf16) is staged once per workgroup into LDS with the gfx1250
// async load-to-LDS path (ASYNCcnt), then each B fragment read from LDS
// feeds two WMMAs (2 A-tiles per wave).
__global__ void __launch_bounds__(256) k_theta(
        const float* __restrict__ xyz, const int* __restrict__ knn,
        const float* __restrict__ lp, const float* __restrict__ w1,
        const float* __restrict__ b2v, const __bf16* __restrict__ w2T,
        const float* __restrict__ w3, const float* __restrict__ b3,
        float* __restrict__ theta_out) {
    __shared__ __bf16 sW2[HHc * HHc];   // 32 KB, [n][k] transposed layout
    __shared__ float  sW1[3 * HHc];     // theta_w1 rows 0..2
    __shared__ float  sLP[HHc];         // lang_part for this batch

    int tid = threadIdx.x;
    int pblock = blockIdx.x * 16;       // 16 consecutive points, all in one batch
    int bIdx = pblock / Nc;

    // --- async DMA of w2T into LDS (256 threads * 16B * 8 iters = 32 KB) ---
    {
        uint64_t gbase = (uint64_t)(uintptr_t)w2T;
        uint32_t lds0  = (uint32_t)(uintptr_t)(&sW2[0]);
#pragma unroll
        for (int i = 0; i < 8; i++) {
            uint32_t off  = (uint32_t)(i * 256 + tid) * 16u;
            uint32_t ldsa = lds0 + off;
            asm volatile("global_load_async_to_lds_b128 %0, %1, %2"
                         :: "v"(ldsa), "v"(off), "s"(gbase) : "memory");
        }
    }
    if (tid < 3 * HHc) sW1[tid] = w1[tid];
    if (tid < HHc)     sLP[tid] = lp[bIdx * HHc + tid];
    asm volatile("s_wait_asynccnt 0x0" ::: "memory");
    __syncthreads();

    int wave = tid >> 5, lane = tid & 31;
    int m = lane & 15, hi = lane >> 4;
    int col = m;

    // --- build A operands for 2 points: h1 = relu(delta @ W1[:3] + lang_part) ---
    v16bf a[2][4];
    int pts[2];
    const float* xb = xyz + (size_t)bIdx * Nc * 3;
#pragma unroll
    for (int t = 0; t < 2; t++) {
        int p = pblock + wave * 2 + t;
        pts[t] = p;
        int n = p - bIdx * Nc;
        int idx = knn[(size_t)p * Kc + m];
        float dx = xb[idx * 3 + 0] - xb[n * 3 + 0];
        float dy = xb[idx * 3 + 1] - xb[n * 3 + 1];
        float dz = xb[idx * 3 + 2] - xb[n * 3 + 2];
#pragma unroll
        for (int s = 0; s < 4; s++) {
            int c0 = a_c0(s, lane);
#pragma unroll
            for (int w = 0; w < 8; w++) {
                int c1 = c0 + w, c2 = c0 + 16 + w;
                float v1 = fmaxf(fmaf(dx, sW1[c1], fmaf(dy, sW1[HHc + c1], fmaf(dz, sW1[2 * HHc + c1], sLP[c1]))), 0.f);
                float v2 = fmaxf(fmaf(dx, sW1[c2], fmaf(dy, sW1[HHc + c2], fmaf(dz, sW1[2 * HHc + c2], sLP[c2]))), 0.f);
                a[t][s][w] = (__bf16)v1; a[t][s][w + 8] = (__bf16)v2;
            }
        }
    }

    // --- GEMM: h2 = relu(h1 @ W2 + b2), dot with w3, per column tile ---
    float acc[2][8] = {{0, 0, 0, 0, 0, 0, 0, 0}, {0, 0, 0, 0, 0, 0, 0, 0}};
#pragma unroll
    for (int ct = 0; ct < 8; ct++) {
        float bias = b2v[ct * 16 + col];
        v8f d0, d1;
#pragma unroll
        for (int r = 0; r < 8; r++) { d0[r] = bias; d1[r] = bias; }
#pragma unroll
        for (int s = 0; s < 4; s++) {
            const __bf16* bp = sW2 + (size_t)(ct * 16 + col) * HHc + s * 32 + (hi << 3);
            v8bf blo = *reinterpret_cast<const v8bf*>(bp), bhi = *reinterpret_cast<const v8bf*>(bp + 16);
            v16bf bmat = __builtin_shufflevector(blo, bhi, 0, 1, 2, 3, 4, 5, 6, 7, 8, 9, 10, 11, 12, 13, 14, 15);  // ds_load, reused twice
            d0 = wmma_bf16(a[0][s], bmat, d0);
            d1 = wmma_bf16(a[1][s], bmat, d1);
        }
        float w3c = w3[ct * 16 + col];
#pragma unroll
        for (int r = 0; r < 8; r++) {
            acc[0][r] += fmaxf(d0[r], 0.f) * w3c;
            acc[1][r] += fmaxf(d1[r], 0.f) * w3c;
        }
    }
#pragma unroll
    for (int t = 0; t < 2; t++) {
#pragma unroll
        for (int r = 0; r < 8; r++) {
            acc[t][r] += __shfl_xor(acc[t][r], 1);
            acc[t][r] += __shfl_xor(acc[t][r], 2);
            acc[t][r] += __shfl_xor(acc[t][r], 4);
            acc[t][r] += __shfl_xor(acc[t][r], 8);
        }
    }
    if (col == 0) {
        float b3v = b3[0];
        int rbase = hi * 8;
#pragma unroll
        for (int t = 0; t < 2; t++)
#pragma unroll
            for (int r = 0; r < 8; r++)
                theta_out[(size_t)pts[t] * Kc + rbase + r] = tanhf(acc[t][r] + b3v) * PI_Fc;
    }
}

// ---------------- cos/sin * neighbor_mask precompute ----------------
__global__ void k_csm(const float* __restrict__ theta, const int* __restrict__ knn,
                      const float* __restrict__ objn, float* __restrict__ cosm,
                      float* __restrict__ sinm) {
    int i = blockIdx.x * 256 + threadIdx.x;
    if (i >= Bc * Nc * Kc) return;
    int p = i / Kc, b = p / Nc;
    float mask = objn[b * Nc + knn[i]];
    float th = theta[i];
    cosm[i] = cosf(th) * mask;
    sinm[i] = sinf(th) * mask;
}

__global__ void k_denom(const int* __restrict__ knn, const float* __restrict__ objn,
                        float* __restrict__ denom) {
    int p = blockIdx.x * 256 + threadIdx.x;
    if (p >= Bc * Nc) return;
    int b = p / Nc;
    float s = 0.f;
    for (int k = 0; k < Kc; k++) s += objn[b * Nc + knn[(size_t)p * Kc + k]];
    denom[p] = fmaxf(s, 1.0f);
}

// ---------------- one diffusion step (called 3x, ping-pong) ----------------
__global__ void k_diffuse(const float* __restrict__ src, float* __restrict__ dst,
                          const int* __restrict__ knn, const float* __restrict__ cosm,
                          const float* __restrict__ sinm, const float* __restrict__ denom) {
    __shared__ int   sidx[4][Kc];
    __shared__ float sc[4][Kc], ss[4][Kc];
    int pl = threadIdx.x >> 6;   // point slot 0..3
    int cp = threadIdx.x & 63;   // channel pair 0..63
    int p = blockIdx.x * 4 + pl;
    int b = p / Nc;
    if (cp < Kc) {
        sidx[pl][cp] = knn[(size_t)p * Kc + cp];
        sc[pl][cp]   = cosm[(size_t)p * Kc + cp];
        ss[pl][cp]   = sinm[(size_t)p * Kc + cp];
    }
    __syncthreads();
    float re = 0.f, im = 0.f;
#pragma unroll
    for (int k = 0; k < Kc; k++) {
        int j = sidx[pl][k];
        float2 f = *reinterpret_cast<const float2*>(src + ((size_t)b * Nc + j) * FEATc + 2 * cp);
        float c = sc[pl][k], s = ss[pl][k];
        re = fmaf(f.x, c, fmaf(-f.y, s, re));
        im = fmaf(f.x, s, fmaf(f.y, c, im));
    }
    float dinv = 1.0f / denom[p];
    float2 u = *reinterpret_cast<const float2*>(src + (size_t)p * FEATc + 2 * cp);
    float2 o;
    o.x = u.x + ALPHAc * (re * dinv - u.x);
    o.y = u.y + ALPHAc * (im * dinv - u.y);
    *reinterpret_cast<float2*>(dst + (size_t)p * FEATc + 2 * cp) = o;
}

// ---------------- LN + proj MLP + score MLP head (WMMA chain) ----------------
__global__ void k_head(const float* __restrict__ updated, const float* __restrict__ features,
                       const float* __restrict__ lang, const float* __restrict__ objn,
                       const float* __restrict__ gamma, const float* __restrict__ beta,
                       const __bf16* __restrict__ pw1T, const float* __restrict__ pb1,
                       const __bf16* __restrict__ pw2T, const float* __restrict__ pb2,
                       const __bf16* __restrict__ sw1T, const float* __restrict__ sb1,
                       const float* __restrict__ sw2, const float* __restrict__ sb2,
                       float* __restrict__ cluster) {
    __shared__ float stage[4][16][FEATc];  // per-wave C->A transpose staging
    int wave = threadIdx.x >> 5, lane = threadIdx.x & 31;
    int tile = blockIdx.x * 4 + wave;
    int row0 = tile * 16;
    int b = row0 / Nc;
    int m = lane & 15;
    int p = row0 + m;  // global point index (== b*N + n)
    const float* urow = updated + (size_t)p * FEATc;

    // LayerNorm: each lane covers its 64 A-layout channels; combine halves via lane^16
    float f[64];
    float sum = 0.f, sq = 0.f;
#pragma unroll
    for (int s = 0; s < 4; s++) {
        int c0 = a_c0(s, lane);
#pragma unroll
        for (int w = 0; w < 8; w++) {
            float v1 = urow[c0 + w], v2 = urow[c0 + 16 + w];
            f[s * 16 + w] = v1; f[s * 16 + 8 + w] = v2;
            sum += v1 + v2; sq += v1 * v1 + v2 * v2;
        }
    }
    sum += __shfl_xor(sum, 16);
    sq  += __shfl_xor(sq, 16);
    float mu = sum * (1.0f / FEATc);
    float var = sq * (1.0f / FEATc) - mu * mu;
    float rstd = rsqrtf(var + LN_EPSc);

    v16bf an[4];
#pragma unroll
    for (int s = 0; s < 4; s++) {
        int c0 = a_c0(s, lane);
#pragma unroll
        for (int w = 0; w < 8; w++) {
            int c1 = c0 + w, c2 = c0 + 16 + w;
            an[s][w]     = (__bf16)((f[s * 16 + w]     - mu) * rstd * gamma[c1] + beta[c1]);
            an[s][w + 8] = (__bf16)((f[s * 16 + 8 + w] - mu) * rstd * gamma[c2] + beta[c2]);
        }
    }

    int col = lane & 15;
    int rbase = (lane >> 4) * 8;

    // t1 = relu(normed @ proj_w1 + b1) -> stage
#pragma unroll
    for (int ct = 0; ct < 8; ct++) {
        float bias = pb1[ct * 16 + col];
        v8f d;
#pragma unroll
        for (int r = 0; r < 8; r++) d[r] = bias;
#pragma unroll
        for (int s = 0; s < 4; s++) d = wmma_bf16(an[s], load_wT_b(pw1T, FEATc, s, ct, lane), d);
#pragma unroll
        for (int r = 0; r < 8; r++) stage[wave][rbase + r][ct * 16 + col] = fmaxf(d[r], 0.f);
    }
    __syncthreads();
    v16bf at[4];
#pragma unroll
    for (int s = 0; s < 4; s++) {
        const float* srow = stage[wave][m];
        int c0 = a_c0(s, lane);
#pragma unroll
        for (int w = 0; w < 8; w++) {
            at[s][w]     = (__bf16)srow[c0 + w];
            at[s][w + 8] = (__bf16)srow[c0 + 16 + w];
        }
    }
    __syncthreads();
    // transported = t1 @ proj_w2 + b2 -> stage
#pragma unroll
    for (int ct = 0; ct < 8; ct++) {
        float bias = pb2[ct * 16 + col];
        v8f d;
#pragma unroll
        for (int r = 0; r < 8; r++) d[r] = bias;
#pragma unroll
        for (int s = 0; s < 4; s++) d = wmma_bf16(at[s], load_wT_b(pw2T, FEATc, s, ct, lane), d);
#pragma unroll
        for (int r = 0; r < 8; r++) stage[wave][rbase + r][ct * 16 + col] = d[r];
    }
    __syncthreads();

    // fused = [transported(0..127), features(128..255), lang(256..511)]; 16 k-steps of 32
    const float* frow = features + (size_t)p * FEATc;
    const float* lrow = lang + (size_t)b * LANGc;
    float acc[8] = {0, 0, 0, 0, 0, 0, 0, 0};
#pragma unroll
    for (int ct = 0; ct < 8; ct++) {
        float bias = sb1[ct * 16 + col];
        v8f d;
#pragma unroll
        for (int r = 0; r < 8; r++) d[r] = bias;
        for (int s = 0; s < 16; s++) {
            const float* seg;
            int coff;
            if (s < 4)       { seg = stage[wave][m]; coff = 0;   }
            else if (s < 8)  { seg = frow;           coff = 128; }
            else             { seg = lrow;           coff = 256; }
            int c0 = a_c0(s, lane) - coff;
            v16bf av;
#pragma unroll
            for (int w = 0; w < 8; w++) {
                av[w]     = (__bf16)seg[c0 + w];
                av[w + 8] = (__bf16)seg[c0 + 16 + w];
            }
            d = wmma_bf16(av, load_wT_b(sw1T, 512, s, ct, lane), d);
        }
        float w2c = sw2[ct * 16 + col];
#pragma unroll
        for (int r = 0; r < 8; r++) acc[r] += fmaxf(d[r], 0.f) * w2c;
    }
#pragma unroll
    for (int r = 0; r < 8; r++) {
        acc[r] += __shfl_xor(acc[r], 1);
        acc[r] += __shfl_xor(acc[r], 2);
        acc[r] += __shfl_xor(acc[r], 4);
        acc[r] += __shfl_xor(acc[r], 8);
    }
    if (col == 0) {
        float bb = sb2[0];
#pragma unroll
        for (int r = 0; r < 8; r++) {
            int pp = row0 + rbase + r;
            cluster[pp] = (acc[r] + bb) * objn[pp];
        }
    }
}

// ---------------- launcher ----------------
extern "C" void kernel_launch(void* const* d_in, const int* in_sizes, int n_in,
                              void* d_out, int out_size, void* d_ws, size_t ws_size,
                              hipStream_t stream) {
    const float* xyz      = (const float*)d_in[0];
    const float* features = (const float*)d_in[1];
    const float* lang     = (const float*)d_in[2];
    const float* objsc    = (const float*)d_in[3];
    const float* theta_w1 = (const float*)d_in[4];
    const float* theta_b1 = (const float*)d_in[5];
    const float* theta_w2 = (const float*)d_in[6];
    const float* theta_b2 = (const float*)d_in[7];
    const float* theta_w3 = (const float*)d_in[8];
    const float* theta_b3 = (const float*)d_in[9];
    const float* ln_gamma = (const float*)d_in[10];
    const float* ln_beta  = (const float*)d_in[11];
    const float* proj_w1  = (const float*)d_in[12];
    const float* proj_b1  = (const float*)d_in[13];
    const float* proj_w2  = (const float*)d_in[14];
    const float* proj_b2  = (const float*)d_in[15];
    const float* score_w1 = (const float*)d_in[16];
    const float* score_b1 = (const float*)d_in[17];
    const float* score_w2 = (const float*)d_in[18];
    const float* score_b2 = (const float*)d_in[19];

    float* out_cluster = (float*)d_out;
    float* out_updated = (float*)d_out + (size_t)Bc * Nc;

    char* w = (char*)d_ws;
    auto alloc = [&](size_t bytes) -> void* {
        void* pp = (void*)w;
        w += (bytes + 255) & ~(size_t)255;
        return pp;
    };
    float*  objn  = (float*)alloc((size_t)Bc * Nc * 4);
    int*    knn   = (int*)  alloc((size_t)Bc * Nc * Kc * 4);
    float*  lp    = (float*)alloc((size_t)Bc * HHc * 4);
    float*  theta = (float*)alloc((size_t)Bc * Nc * Kc * 4);
    float*  cosm  = (float*)alloc((size_t)Bc * Nc * Kc * 4);
    float*  sinm  = (float*)alloc((size_t)Bc * Nc * Kc * 4);
    float*  denom = (float*)alloc((size_t)Bc * Nc * 4);
    float*  Ua    = (float*)alloc((size_t)Bc * Nc * FEATc * 4);
    float*  Ub    = (float*)alloc((size_t)Bc * Nc * FEATc * 4);
    __bf16* tw2T  = (__bf16*)alloc((size_t)HHc * HHc * 2);
    __bf16* pw1T  = (__bf16*)alloc((size_t)FEATc * HIDc * 2);
    __bf16* pw2T  = (__bf16*)alloc((size_t)HIDc * FEATc * 2);
    __bf16* sw1T  = (__bf16*)alloc((size_t)512 * HIDc * 2);

    // phase 0: objectness + weight prep
    k_objn<<<(Bc * Nc + 255) / 256, 256, 0, stream>>>(objsc, objn);
    k_prep_wT<<<(HHc * HHc + 255) / 256, 256, 0, stream>>>(theta_w2, tw2T, HHc, HHc);
    k_prep_wT<<<(FEATc * HIDc + 255) / 256, 256, 0, stream>>>(proj_w1, pw1T, FEATc, HIDc);
    k_prep_wT<<<(HIDc * FEATc + 255) / 256, 256, 0, stream>>>(proj_w2, pw2T, HIDc, FEATc);
    k_prep_wT<<<(512 * HIDc + 255) / 256, 256, 0, stream>>>(score_w1, sw1T, 512, HIDc);

    // phase 1: KNN
    {
        dim3 grid(Nc / 256, Bc);
        k_knn<<<grid, 256, 0, stream>>>(xyz, knn);
    }

    // phase 2: theta MLP (WMMA, LDS-staged weights via async-to-LDS) + rotation tables
    k_langpart<<<Bc, HHc, 0, stream>>>(lang, theta_w1, theta_b1, lp);
    k_theta<<<(Bc * Nc) / 16, 256, 0, stream>>>(xyz, knn, lp, theta_w1, theta_b2, tw2T,
                                                theta_w3, theta_b3, theta);
    k_csm<<<(Bc * Nc * Kc + 255) / 256, 256, 0, stream>>>(theta, knn, objn, cosm, sinm);
    k_denom<<<(Bc * Nc + 255) / 256, 256, 0, stream>>>(knn, objn, denom);

    // phase 3: 3 diffusion steps (features -> Ua -> Ub -> out_updated)
    k_diffuse<<<(Bc * Nc) / 4, 256, 0, stream>>>(features, Ua, knn, cosm, sinm, denom);
    k_diffuse<<<(Bc * Nc) / 4, 256, 0, stream>>>(Ua, Ub, knn, cosm, sinm, denom);
    k_diffuse<<<(Bc * Nc) / 4, 256, 0, stream>>>(Ub, out_updated, knn, cosm, sinm, denom);

    // phase 4: LN + proj MLP + score MLP (WMMA chain), conf * objn
    k_head<<<(Bc * Nc) / 64, 128, 0, stream>>>(out_updated, features, lang, objn,
                                               ln_gamma, ln_beta, pw1T, proj_b1, pw2T, proj_b2,
                                               sw1T, score_b1, score_w2, score_b2, out_cluster);
}
